// WaveNeuralNetwork_91164975825525
// MI455X (gfx1250) — matrix operand order repaired and checked by reading, not compiled, not run on hardware
//
#include <hip/hip_runtime.h>

typedef _Float16 half_t;
typedef __attribute__((ext_vector_type(16))) _Float16 v16h;
typedef __attribute__((ext_vector_type(8)))  _Float16 v8h;
typedef __attribute__((ext_vector_type(8)))  float    v8f;

namespace {
constexpr int kB  = 1024;   // batch
constexpr int kT  = 1024;   // time steps
constexpr int kH  = 97;     // hidden
constexpr int kO  = 2;      // outputs
constexpr int kTB = 16;     // time-block = WMMA M dimension
constexpr int kRow = 136;   // padded LDS row (halves): h 0..127 used, padded for banks/align
}

// Uniform lane broadcast via v_readlane (SGPR result, no LDS round-trip).
static __device__ __forceinline__ float lane_bcast(float v, int l) {
  return __uint_as_float(__builtin_amdgcn_readlane(__float_as_uint(v), l));
}

// One wave32 per batch element.  Blocks time by 16:
//   phase 1: 16 steps of the layer-1 LIF recurrence (lane owns h = lane+32j),
//            coalesced NT stores of spk1/mem1, spikes staged in LDS as f16.
//   phase 2: cur2 = spk1(16x97) @ W2^T(97x2) via v_wmma_f32_16x16x32_f16
//            (K padded 97->128, 4 chained WMMAs), then the 16-step mem2
//            recurrence unrolled from the D accumulator via v_readlane.
__global__ __launch_bounds__(32) void snn_fused(
    const float* __restrict__ x,
    const float* __restrict__ W1, const float* __restrict__ b1,
    const float* __restrict__ beta1, const float* __restrict__ thr1,
    const float* __restrict__ W2, const float* __restrict__ b2,
    const float* __restrict__ beta2, const float* __restrict__ thr2,
    float* __restrict__ out)
{
  __shared__ __align__(16) half_t tile[kTB][kRow];   // per-wave private (1-wave WG)

  const int lane = threadIdx.x;
  const int b    = blockIdx.x;

  // ---- layer-1 params for this lane's hidden units h = lane + 32j ----
  float w1h[4], b1h[4], be1[4], th1[4], mem1[4];
#pragma unroll
  for (int j = 0; j < 4; ++j) {
    int h  = lane + 32 * j;
    int hc = (h < kH) ? h : (kH - 1);      // clamp; h>=97 lanes masked on store, W2 col = 0
    w1h[j] = W1[hc];
    b1h[j] = b1[hc];
    be1[j] = fminf(fmaxf(beta1[hc], 0.f), 1.f);
    th1[j] = thr1[hc];
    mem1[j] = 0.f;
  }
  const float b2v[2] = { b2[0], b2[1] };
  const float be2[2] = { fminf(fmaxf(beta2[0], 0.f), 1.f),
                         fminf(fmaxf(beta2[1], 0.f), 1.f) };
  const float th2[2] = { thr2[0], thr2[1] };
  float mem2[2] = { 0.f, 0.f };

  // ---- WMMA B operand: W2^T in f16, 4 K-chunks of 32 covering h 0..127 ----
  // B(32x16) in this code's k order (same as A): half i of lane l holds B[k][n], n = l%16, k = 16*(l/16)+i.
  const int n_col = lane & 15;
  const int khalf = lane >> 4;
  v16h bmat[4];
#pragma unroll
  for (int c = 0; c < 4; ++c) {
#pragma unroll
    for (int i = 0; i < 16; ++i) {
      int h = 32 * c + 16 * khalf + i;
      float wv = (n_col < kO && h < kH) ? W2[n_col * kH + h] : 0.f;
      bmat[c][i] = (half_t)wv;
    }
  }

  const int TBH = kT * kB * kH;
  const int TBO = kT * kB * kO;
  float* __restrict__ spk1_rec = out;
  float* __restrict__ mem1_rec = out + TBH;
  float* __restrict__ spk2_rec = out + 2 * TBH;
  float* __restrict__ mem2_rec = out + 2 * TBH + TBO;

  const int my_m = lane >> 1;   // this lane stores layer-2 (t_local = my_m, o = my_o)
  const int my_o = lane & 1;

#pragma unroll 1
  for (int t0 = 0; t0 < kT; t0 += kTB) {
    // 16 x-values of this block: lane i holds x[b][t0+i] (i<16), broadcast via readlane
    float xv = x[b * kT + t0 + (lane & 15)];

    // ================= phase 1: 16 steps of layer 1 =================
#pragma unroll 4
    for (int tl = 0; tl < kTB; ++tl) {
      float xt = lane_bcast(xv, tl);
      int base1 = (t0 + tl) * (kB * kH) + b * kH;
#pragma unroll
      for (int j = 0; j < 4; ++j) {
        int h = lane + 32 * j;
        float cur = fmaf(xt, w1h[j], b1h[j]);
        float rst = (mem1[j] > th1[j]) ? th1[j] : 0.f;   // reset from pre-update mem
        float m   = fmaf(be1[j], mem1[j], cur) - rst;
        mem1[j]   = m;
        float spk = ((m - th1[j]) > 0.f) ? 1.f : 0.f;
        tile[tl][h] = (half_t)spk;                        // 0/1 exact in f16
        if (h < kH) {
          __builtin_nontemporal_store(spk, spk1_rec + base1 + h);  // coalesced 128B
          __builtin_nontemporal_store(m,   mem1_rec + base1 + h);
        }
      }
    }
    __syncthreads();   // single-wave WG: just orders DS ops (barrier is a NOP)

    // ====== phase 2: cur2 = spk1(16x97) @ W2^T via v_wmma_f32_16x16x32_f16 ======
    v8f acc = {};
#pragma unroll
    for (int c = 0; c < 4; ++c) {
      const int m = lane & 15;   // A row = t_local; lane-half selects K groups
      v8h a0 = *(const v8h*)&tile[m][32 * c + 16 * khalf];       // K = 16*khalf+0..7
      v8h a1 = *(const v8h*)&tile[m][32 * c + 16 * khalf + 8];   // K = 16*khalf+8..15
      v16h amat;
#pragma unroll
      for (int i = 0; i < 8; ++i) { amat[i] = a0[i]; amat[8 + i] = a1[i]; }
      acc = __builtin_amdgcn_wmma_f32_16x16x32_f16(
          false, amat, false, bmat[c], (short)0, acc, false, false);
    }

    // ---- layer-2 recurrence over the 16 steps (all lanes redundant, uniform) ----
    float spk2_out = 0.f, mem2_out = 0.f;
#pragma unroll
    for (int m = 0; m < 16; ++m) {
      const int srcl = (m < 8) ? 0 : 16;          // D: VGPR m%8 holds M=m per lane-half
      float c0 = lane_bcast(acc[m & 7], srcl + 0) + b2v[0];
      float c1 = lane_bcast(acc[m & 7], srcl + 1) + b2v[1];
      float r0 = (mem2[0] > th2[0]) ? th2[0] : 0.f;
      float r1 = (mem2[1] > th2[1]) ? th2[1] : 0.f;
      mem2[0] = fmaf(be2[0], mem2[0], c0) - r0;
      mem2[1] = fmaf(be2[1], mem2[1], c1) - r1;
      float s0 = ((mem2[0] - th2[0]) > 0.f) ? 1.f : 0.f;
      float s1 = ((mem2[1] - th2[1]) > 0.f) ? 1.f : 0.f;
      if (my_m == m) {
        spk2_out = my_o ? s1 : s0;
        mem2_out = my_o ? mem2[1] : mem2[0];
      }
    }
    int idx2 = (t0 + my_m) * (kB * kO) + b * kO + my_o;
    __builtin_nontemporal_store(spk2_out, spk2_rec + idx2);
    __builtin_nontemporal_store(mem2_out, mem2_rec + idx2);
    __syncthreads();   // tile reads done before next block overwrites it
  }
}

extern "C" void kernel_launch(void* const* d_in, const int* in_sizes, int n_in,
                              void* d_out, int out_size, void* d_ws, size_t ws_size,
                              hipStream_t stream) {
  (void)in_sizes; (void)n_in; (void)out_size; (void)d_ws; (void)ws_size;
  const float* x     = (const float*)d_in[0];
  const float* W1    = (const float*)d_in[1];
  const float* b1    = (const float*)d_in[2];
  const float* beta1 = (const float*)d_in[3];
  const float* thr1  = (const float*)d_in[4];
  const float* W2    = (const float*)d_in[5];
  const float* b2    = (const float*)d_in[6];
  const float* beta2 = (const float*)d_in[7];
  const float* thr2  = (const float*)d_in[8];
  snn_fused<<<dim3(kB), dim3(32), 0, stream>>>(
      x, W1, b1, beta1, thr1, W2, b2, beta2, thr2, (float*)d_out);
}
